// RNG_QP_Net_52828097741348
// MI455X (gfx1250) — matrix operand order repaired and checked by reading, not compiled, not run on hardware
//
#include <hip/hip_runtime.h>
#include <hip/hip_bf16.h>
#include <math.h>

// ---------------------------------------------------------------------------
// MI455X (gfx1250, wave32) implementation of the PointNet+QP reference.
// Heavy per-point GEMMs and Gram-matrix (covariance) updates use
// v_wmma_f32_16x16x32_f16 (f16 A/B, fp32 accumulate). BatchNorm statistics are
// derived from activation means + Gram matrices (linear-layer trick), so no
// pre-BN activation tensor is ever materialized in HBM.
// ---------------------------------------------------------------------------

typedef __attribute__((ext_vector_type(16))) _Float16 v16h;
typedef __attribute__((ext_vector_type(8)))  _Float16 v8h;
typedef __attribute__((ext_vector_type(8)))  float    v8f;

#define NBATCH   64
#define NPTS     8192
#define PTOT     (NBATCH * NPTS)      // 524288 points
#define CHUNK    128                  // points per block iteration
#define NCHUNK   (PTOT / CHUNK)       // 4096
#define NB_STAT  256                  // blocks for stats kernels
#define BN_EPS_F 1e-5f

union Vh16 { v16h v; v8h h[2]; _Float16 e[16]; };

__device__ __forceinline__ v8f zero8() {
  v8f z = {0.f,0.f,0.f,0.f,0.f,0.f,0.f,0.f};
  return z;
}

#define WMMA_F16(A,B,C) \
  __builtin_amdgcn_wmma_f32_16x16x32_f16(false,(A),false,(B),(short)0,(C),false,false)

// A-fragment (16x32, f16): rows (M) indexed by lane&15 (+row0),
//   K = k0 + 16*half + e
// => two contiguous 16B LDS loads per lane.
__device__ __forceinline__ v16h fragA_lds(const _Float16* buf, int stride, int row0, int k0) {
  int lane = threadIdx.x & 31;
  int half = lane >> 4;
  const _Float16* p = buf + (size_t)(row0 + (lane & 15)) * stride + k0 + 16 * half;
  Vh16 r;
  r.h[0] = *(const v8h*)p;
  r.h[1] = *(const v8h*)(p + 8);
  return r.v;
}

// B-fragment (32x16, f16): cols (N) indexed by lane&15 (+row0 = column base),
// K = k0 + 16*half + e => one contiguous 32B run per lane.
__device__ __forceinline__ v16h fragB_lds(const _Float16* buf, int stride, int row0, int k0) {
  int lane = threadIdx.x & 31;
  int half = lane >> 4;
  const _Float16* p = buf + (size_t)(row0 + (lane & 15)) * stride + k0 + 16 * half;
  Vh16 r;
  r.h[0] = *(const v8h*)p;
  r.h[1] = *(const v8h*)(p + 8);
  return r.v;
}

// B-fragment from a row-major fp32 weight matrix W[c][k] (B[k][c] = W[c][k]).
__device__ __forceinline__ v16h fragB_wt(const float* W, int ld, int c0, int k0) {
  int lane = threadIdx.x & 31;
  int half = lane >> 4;
  const float* p = W + (size_t)(c0 + (lane & 15)) * ld + k0 + 16 * half;
  const float4* p4 = (const float4*)p;
  Vh16 r;
#pragma unroll
  for (int q = 0; q < 4; ++q) {
    float4 f = p4[q];
    r.e[4*q+0] = (_Float16)f.x; r.e[4*q+1] = (_Float16)f.y;
    r.e[4*q+2] = (_Float16)f.z; r.e[4*q+3] = (_Float16)f.w;
  }
  return r.v;
}

// ---------------------------------------------------------------------------
// K0: first/second moments of pcd (3 sums + 6 second moments, fp32, fixed order)
// ---------------------------------------------------------------------------
__global__ __launch_bounds__(256) void k0_pcd_stats(const float* __restrict__ pcd,
                                                    float* __restrict__ part) {
  float a[9] = {0,0,0,0,0,0,0,0,0};
  for (int i = blockIdx.x * 256 + threadIdx.x; i < PTOT; i += 256 * gridDim.x) {
    float x = pcd[3*i], y = pcd[3*i+1], z = pcd[3*i+2];
    a[0] += x;   a[1] += y;   a[2] += z;
    a[3] += x*x; a[4] += y*y; a[5] += z*z;
    a[6] += x*y; a[7] += x*z; a[8] += y*z;
  }
  __shared__ float red[256];
#pragma unroll
  for (int q = 0; q < 9; ++q) {
    red[threadIdx.x] = a[q];
    __syncthreads();
    for (int off = 128; off > 0; off >>= 1) {
      if (threadIdx.x < (unsigned)off) red[threadIdx.x] += red[threadIdx.x + off];
      __syncthreads();
    }
    if (threadIdx.x == 0) part[blockIdx.x * 9 + q] = red[0];
    __syncthreads();
  }
}

// K0b: reduce pcd moments, derive BN1 scale/shift analytically:
// mu1[c] = W1[c]·mean + b1 ; var1[c] = W1[c]·Cov·W1[c]^T
__global__ void k0b_bn1(const float* __restrict__ part, int nparts,
                        const float* __restrict__ W1, const float* __restrict__ b1,
                        const float* __restrict__ g1, const float* __restrict__ be1,
                        float* __restrict__ s1, float* __restrict__ t1) {
  __shared__ float st[9];
  int t = threadIdx.x;
  if (t < 9) {
    float s = 0.f;
    for (int p = 0; p < nparts; ++p) s += part[p * 9 + t];
    st[t] = s;
  }
  __syncthreads();
  if (t < 64) {
    const float inv = 1.f / (float)PTOT;
    float m0 = st[0]*inv, m1 = st[1]*inv, m2 = st[2]*inv;
    float C[3][3];
    C[0][0] = st[3]*inv - m0*m0; C[1][1] = st[4]*inv - m1*m1; C[2][2] = st[5]*inv - m2*m2;
    C[0][1] = C[1][0] = st[6]*inv - m0*m1;
    C[0][2] = C[2][0] = st[7]*inv - m0*m2;
    C[1][2] = C[2][1] = st[8]*inv - m1*m2;
    float w[3] = {W1[3*t], W1[3*t+1], W1[3*t+2]};
    float mu = b1[t] + w[0]*m0 + w[1]*m1 + w[2]*m2;
    float var = 0.f;
#pragma unroll
    for (int j = 0; j < 3; ++j)
#pragma unroll
      for (int k = 0; k < 3; ++k) var += w[j] * w[k] * C[j][k];
    float sc = g1[t] * rsqrtf(var + BN_EPS_F);
    s1[t] = sc;
    t1[t] = be1[t] - mu * sc;
  }
}

// ---------------------------------------------------------------------------
// K1: stream points, a1 = relu(bn1(pcd@W1^T+b1)); accumulate sum(a1)[64] and
//     S1 = a1^T a1 (64x64 Gram) with WMMA. a1 stored channel-major in LDS.
// ---------------------------------------------------------------------------
__global__ __launch_bounds__(256) void k1_gram1(const float* __restrict__ pcd,
                                                const float* __restrict__ W1,
                                                const float* __restrict__ b1,
                                                const float* __restrict__ s1,
                                                const float* __restrict__ t1,
                                                float* __restrict__ part) {
  __shared__ __align__(16) _Float16 a1T[64 * CHUNK];   // [ch][pt]
  __shared__ float pcdS[CHUNK * 3];
  __shared__ float msred[256];
  int t = threadIdx.x, w = t >> 5, lane = t & 31;
  int ch = t & 63, g = t >> 6;
  float w0 = W1[3*ch], w1 = W1[3*ch+1], w2 = W1[3*ch+2];
  float bb = b1[ch], ss = s1[ch], tt = t1[ch];
  v8f Sacc[2]; Sacc[0] = zero8(); Sacc[1] = zero8();
  float msum = 0.f;

  for (int ck = blockIdx.x; ck < NCHUNK; ck += gridDim.x) {
    __syncthreads();
    if (t < CHUNK) {
      int base = (ck * CHUNK + t) * 3;
      pcdS[3*t] = pcd[base]; pcdS[3*t+1] = pcd[base+1]; pcdS[3*t+2] = pcd[base+2];
    }
    __syncthreads();
    for (int i = 0; i < 32; ++i) {
      int pt = g * 32 + i;
      float x0 = pcdS[3*pt], x1 = pcdS[3*pt+1], x2 = pcdS[3*pt+2];
      float y = fmaf(x0, w0, fmaf(x1, w1, fmaf(x2, w2, bb)));
      float a = fmaxf(fmaf(y, ss, tt), 0.f);
      msum += a;
      a1T[ch * CHUNK + pt] = (_Float16)a;
    }
    __syncthreads();
    // S1 += a1^T a1 : wave w owns tiles 2w, 2w+1 of the 4x4 tile grid
#pragma unroll
    for (int i = 0; i < 2; ++i) {
      int tile = 2 * w + i;
      int cm0 = (tile >> 2) * 16, cn0 = (tile & 3) * 16;
#pragma unroll
      for (int ks = 0; ks < 4; ++ks) {
        v16h Af = fragA_lds(a1T, CHUNK, cm0, 32 * ks);
        v16h Bf = fragB_lds(a1T, CHUNK, cn0, 32 * ks);
        Sacc[i] = WMMA_F16(Af, Bf, Sacc[i]);
      }
    }
  }
  float* S1p = part + (size_t)blockIdx.x * 4160;
#pragma unroll
  for (int i = 0; i < 2; ++i) {
    int tile = 2 * w + i;
    int cm0 = (tile >> 2) * 16, cn0 = (tile & 3) * 16;
    int col = cn0 + (lane & 15), half = lane >> 4;
#pragma unroll
    for (int v = 0; v < 8; ++v) {
      int row = cm0 + v + 8 * half;
      S1p[row * 64 + col] = Sacc[i][v];
    }
  }
  msred[t] = msum;
  __syncthreads();
  if (t < 64) S1p[4096 + t] = msred[t] + msred[64 + t] + msred[128 + t] + msred[192 + t];
}

// Generic fixed-order partial reducer
__global__ void k_reduce(const float* __restrict__ src, float* __restrict__ dst,
                         int count, int nparts) {
  int i = blockIdx.x * blockDim.x + threadIdx.x;
  if (i >= count) return;
  float s = 0.f;
  for (int p = 0; p < nparts; ++p) s += src[(size_t)p * count + i];
  dst[i] = s;
}

// K1c: BN2 params from mean(a1) and Gram S1.  var2[c]=W2[c]·Cov1·W2[c]^T
__global__ void k1c_bn2(const float* __restrict__ S1red, const float* __restrict__ W2,
                        const float* __restrict__ b2, const float* __restrict__ g2,
                        const float* __restrict__ be2,
                        float* __restrict__ s2, float* __restrict__ t2) {
  __shared__ float m[64];
  int c = threadIdx.x;                      // 128 threads
  const float inv = 1.f / (float)PTOT;
  if (c < 64) m[c] = S1red[4096 + c] * inv;
  __syncthreads();
  const float* Wr = W2 + (size_t)c * 64;
  float mu = b2[c];
  for (int k = 0; k < 64; ++k) mu = fmaf(Wr[k], m[k], mu);
  float var = 0.f;
  for (int j = 0; j < 64; ++j) {
    float tj = 0.f;
    float mj = m[j];
    for (int k = 0; k < 64; ++k) tj = fmaf(Wr[k], S1red[j*64+k]*inv - mj*m[k], tj);
    var = fmaf(Wr[j], tj, var);
  }
  float sc = g2[c] * rsqrtf(var + BN_EPS_F);
  s2[c] = sc;
  t2[c] = be2[c] - mu * sc;
}

// ---------------------------------------------------------------------------
// K2: recompute a1, layer-2 WMMA GEMM, a2 = relu(bn2(.)); accumulate sum(a2)
//     and S2 = a2^T a2 (128x128) with WMMA. a2 stored channel-major in LDS.
// ---------------------------------------------------------------------------
__global__ __launch_bounds__(256) void k2_gram2(const float* __restrict__ pcd,
                                                const float* __restrict__ W1,
                                                const float* __restrict__ b1,
                                                const float* __restrict__ s1,
                                                const float* __restrict__ t1,
                                                const float* __restrict__ W2,
                                                const float* __restrict__ b2,
                                                const float* __restrict__ s2,
                                                const float* __restrict__ t2,
                                                float* __restrict__ part) {
  __shared__ __align__(16) _Float16 a1pm[CHUNK * 64];    // [pt][ch]
  __shared__ __align__(16) _Float16 a2T[128 * CHUNK];    // [ch][pt]
  __shared__ float pcdS[CHUNK * 3];
  __shared__ float p1S[64 * 6];
  int t = threadIdx.x, w = t >> 5, lane = t & 31, half = lane >> 4;

  v16h Bw0 = fragB_wt(W2, 64, 16 * w, 0);
  v16h Bw1 = fragB_wt(W2, 64, 16 * w, 32);
  int c2 = 16 * w + (lane & 15);
  float s2c = s2[c2];
  float t2c = t2[c2] + b2[c2] * s2c;       // fold layer-2 bias into BN shift
  if (t < 64) {
    p1S[6*t+0] = W1[3*t]; p1S[6*t+1] = W1[3*t+1]; p1S[6*t+2] = W1[3*t+2];
    p1S[6*t+3] = b1[t];   p1S[6*t+4] = s1[t];     p1S[6*t+5] = t1[t];
  }
  v8f Sacc[8];
#pragma unroll
  for (int i = 0; i < 8; ++i) Sacc[i] = zero8();
  float msum = 0.f;

  for (int ck = blockIdx.x; ck < NCHUNK; ck += gridDim.x) {
    __syncthreads();
    if (t < CHUNK) {
      int base = (ck * CHUNK + t) * 3;
      pcdS[3*t] = pcd[base]; pcdS[3*t+1] = pcd[base+1]; pcdS[3*t+2] = pcd[base+2];
    }
    __syncthreads();
    {   // layer 1 (point-major)
      int pt = t >> 1;
      float x0 = pcdS[3*pt], x1 = pcdS[3*pt+1], x2 = pcdS[3*pt+2];
      int c0 = (t & 1) * 32;
      for (int i = 0; i < 32; ++i) {
        int ch = c0 + i;
        const float* p = p1S + 6 * ch;
        float y = fmaf(x0, p[0], fmaf(x1, p[1], fmaf(x2, p[2], p[3])));
        a1pm[pt * 64 + ch] = (_Float16)fmaxf(fmaf(y, p[4], p[5]), 0.f);
      }
    }
    __syncthreads();
    // layer-2 forward: wave w computes channel tile w for all 8 point tiles
#pragma unroll
    for (int pp = 0; pp < 8; ++pp) {
      v8f acc = zero8();
      acc = WMMA_F16(fragA_lds(a1pm, 64, 16 * pp, 0),  Bw0, acc);
      acc = WMMA_F16(fragA_lds(a1pm, 64, 16 * pp, 32), Bw1, acc);
      v8h outv;
#pragma unroll
      for (int v = 0; v < 8; ++v) {
        float f = fmaxf(fmaf(acc[v], s2c, t2c), 0.f);
        msum += f;
        outv[v] = (_Float16)f;
      }
      *(v8h*)(a2T + (size_t)c2 * CHUNK + 16 * pp + 8 * half) = outv;
    }
    __syncthreads();
    // S2 += a2^T a2 : wave w owns row-block w (tiles (w, 0..7))
#pragma unroll
    for (int ks = 0; ks < 4; ++ks) {
      v16h Af = fragA_lds(a2T, CHUNK, 16 * w, 32 * ks);
#pragma unroll
      for (int i = 0; i < 8; ++i) {
        v16h Bf = fragB_lds(a2T, CHUNK, 16 * i, 32 * ks);
        Sacc[i] = WMMA_F16(Af, Bf, Sacc[i]);
      }
    }
  }
  float* S2p = part + (size_t)blockIdx.x * 16512;
#pragma unroll
  for (int i = 0; i < 8; ++i) {
    int col = 16 * i + (lane & 15);
#pragma unroll
    for (int v = 0; v < 8; ++v) {
      int row = 16 * w + v + 8 * half;
      S2p[row * 128 + col] = Sacc[i][v];
    }
  }
  float other = __shfl(msum, lane ^ 16);
  if (lane < 16) S2p[16384 + 16 * w + lane] = msum + other;
}

// K2c: BN3 params from mean(a2) and Gram S2
__global__ void k2c_bn3(const float* __restrict__ S2red, const float* __restrict__ W3,
                        const float* __restrict__ b3, const float* __restrict__ g3,
                        const float* __restrict__ be3,
                        float* __restrict__ s3, float* __restrict__ t3) {
  __shared__ float m[128];
  int c = threadIdx.x;                     // 256 threads
  const float inv = 1.f / (float)PTOT;
  if (c < 128) m[c] = S2red[16384 + c] * inv;
  __syncthreads();
  const float* Wr = W3 + (size_t)c * 128;
  float mu = b3[c];
  for (int k = 0; k < 128; ++k) mu = fmaf(Wr[k], m[k], mu);
  float var = 0.f;
  for (int j = 0; j < 128; ++j) {
    float tj = 0.f;
    float mj = m[j];
    for (int k = 0; k < 128; ++k) tj = fmaf(Wr[k], S2red[j*128+k]*inv - mj*m[k], tj);
    var = fmaf(Wr[j], tj, var);
  }
  float sc = g3[c] * rsqrtf(var + BN_EPS_F);
  s3[c] = sc;
  t3[c] = be3[c] - mu * sc;
}

// ---------------------------------------------------------------------------
// K3: full forward (a1 VALU, a2 WMMA, a3 WMMA) fused with BN3+ReLU+per-batch
//     max over points. One block per (batch, slice-of-512-points).
// ---------------------------------------------------------------------------
__global__ __launch_bounds__(256) void k3_forward_max(const float* __restrict__ pcd,
                                                      const float* __restrict__ W1,
                                                      const float* __restrict__ b1,
                                                      const float* __restrict__ s1,
                                                      const float* __restrict__ t1,
                                                      const float* __restrict__ W2,
                                                      const float* __restrict__ b2,
                                                      const float* __restrict__ s2,
                                                      const float* __restrict__ t2,
                                                      const float* __restrict__ W3,
                                                      const float* __restrict__ b3,
                                                      const float* __restrict__ s3,
                                                      const float* __restrict__ t3,
                                                      float* __restrict__ maxp) {
  __shared__ __align__(16) _Float16 a1pm[CHUNK * 64];    // [pt][ch]
  __shared__ __align__(16) _Float16 a2pm[CHUNK * 128];   // [pt][ch]
  __shared__ float pcdS[CHUNK * 3];
  __shared__ float p1S[64 * 6];
  int t = threadIdx.x, w = t >> 5, lane = t & 31, half = lane >> 4;
  int bb = blockIdx.x, b = bb >> 4, sl = bb & 15;

  v16h Bw20 = fragB_wt(W2, 64, 16 * w, 0);
  v16h Bw21 = fragB_wt(W2, 64, 16 * w, 32);
  int c2 = 16 * w + (lane & 15);
  float s2c = s2[c2];
  float t2c = t2[c2] + b2[c2] * s2c;

  v16h Bw3[2][4];
  float s3c[2], t3c[2];
#pragma unroll
  for (int j = 0; j < 2; ++j) {
    int c3 = 16 * (2 * w + j) + (lane & 15);
    s3c[j] = s3[c3];
    t3c[j] = t3[c3] + b3[c3] * s3c[j];
#pragma unroll
    for (int ks = 0; ks < 4; ++ks) Bw3[j][ks] = fragB_wt(W3, 128, 16 * (2 * w + j), 32 * ks);
  }
  if (t < 64) {
    p1S[6*t+0] = W1[3*t]; p1S[6*t+1] = W1[3*t+1]; p1S[6*t+2] = W1[3*t+2];
    p1S[6*t+3] = b1[t];   p1S[6*t+4] = s1[t];     p1S[6*t+5] = t1[t];
  }
  float rmax[2] = {-3.4e38f, -3.4e38f};

  for (int ckl = 0; ckl < 4; ++ckl) {
    int base_pt = b * NPTS + sl * 512 + ckl * CHUNK;
    __syncthreads();
    if (t < CHUNK) {
      int base = (base_pt + t) * 3;
      pcdS[3*t] = pcd[base]; pcdS[3*t+1] = pcd[base+1]; pcdS[3*t+2] = pcd[base+2];
    }
    __syncthreads();
    {   // layer 1
      int pt = t >> 1;
      float x0 = pcdS[3*pt], x1 = pcdS[3*pt+1], x2 = pcdS[3*pt+2];
      int c0 = (t & 1) * 32;
      for (int i = 0; i < 32; ++i) {
        int ch = c0 + i;
        const float* p = p1S + 6 * ch;
        float y = fmaf(x0, p[0], fmaf(x1, p[1], fmaf(x2, p[2], p[3])));
        a1pm[pt * 64 + ch] = (_Float16)fmaxf(fmaf(y, p[4], p[5]), 0.f);
      }
    }
    __syncthreads();
    // layer-2 forward, point-major output
#pragma unroll
    for (int pp = 0; pp < 8; ++pp) {
      v8f acc = zero8();
      acc = WMMA_F16(fragA_lds(a1pm, 64, 16 * pp, 0),  Bw20, acc);
      acc = WMMA_F16(fragA_lds(a1pm, 64, 16 * pp, 32), Bw21, acc);
#pragma unroll
      for (int v = 0; v < 8; ++v) {
        float f = fmaxf(fmaf(acc[v], s2c, t2c), 0.f);
        a2pm[(size_t)(16 * pp + v + 8 * half) * 128 + c2] = (_Float16)f;
      }
    }
    __syncthreads();
    // layer-3 forward fused with BN3+ReLU+max; wave w owns channel tiles 2w,2w+1
#pragma unroll
    for (int pp = 0; pp < 8; ++pp) {
      v16h A0 = fragA_lds(a2pm, 128, 16 * pp, 0);
      v16h A1 = fragA_lds(a2pm, 128, 16 * pp, 32);
      v16h A2 = fragA_lds(a2pm, 128, 16 * pp, 64);
      v16h A3 = fragA_lds(a2pm, 128, 16 * pp, 96);
#pragma unroll
      for (int j = 0; j < 2; ++j) {
        v8f acc = zero8();
        acc = WMMA_F16(A0, Bw3[j][0], acc);
        acc = WMMA_F16(A1, Bw3[j][1], acc);
        acc = WMMA_F16(A2, Bw3[j][2], acc);
        acc = WMMA_F16(A3, Bw3[j][3], acc);
#pragma unroll
        for (int v = 0; v < 8; ++v) {
          float f = fmaxf(fmaf(acc[v], s3c[j], t3c[j]), 0.f);
          rmax[j] = fmaxf(rmax[j], f);
        }
      }
    }
  }
#pragma unroll
  for (int j = 0; j < 2; ++j) {
    float o = __shfl(rmax[j], lane ^ 16);
    float m = fmaxf(rmax[j], o);
    if (lane < 16) maxp[(size_t)bb * 256 + 32 * w + 16 * j + lane] = m;
  }
}

// K3b: reduce per-slice maxes, apply fc (256->128)
__global__ void k3b_feats(const float* __restrict__ maxp, const float* __restrict__ Wfc,
                          const float* __restrict__ bfc, float* __restrict__ feats) {
  __shared__ float mx[256];
  int b = blockIdx.x, t = threadIdx.x;
  float m = -3.4e38f;
  for (int s = 0; s < 16; ++s) m = fmaxf(m, maxp[(size_t)(b * 16 + s) * 256 + t]);
  mx[t] = m;
  __syncthreads();
  if (t < 128) {
    float acc = bfc[t];
    const float* Wr = Wfc + (size_t)t * 256;
    for (int k = 0; k < 256; ++k) acc = fmaf(Wr[k], mx[k], acc);
    feats[b * 128 + t] = acc;
  }
}

// K4: heads (p_pred, normals, h) — tiny, one block per batch element
__global__ void k4_heads(const float* __restrict__ feats,
                         const float* __restrict__ Wc1, const float* __restrict__ bc1,
                         const float* __restrict__ Wc2, const float* __restrict__ bc2,
                         const float* __restrict__ Wp1, const float* __restrict__ bp1,
                         const float* __restrict__ Wp2, const float* __restrict__ bp2,
                         float* __restrict__ out) {
  __shared__ float fS[128], h1[128], hp[64], oc[48];
  int b = blockIdx.x, t = threadIdx.x;          // 128 threads
  fS[t] = feats[b * 128 + t];
  __syncthreads();
  {
    float acc = bc1[t];
    const float* Wr = Wc1 + (size_t)t * 128;
    for (int k = 0; k < 128; ++k) acc = fmaf(Wr[k], fS[k], acc);
    h1[t] = fmaxf(acc, 0.f);
  }
  if (t < 64) {
    float acc = bp1[t];
    const float* Wr = Wp1 + (size_t)t * 128;
    for (int k = 0; k < 128; ++k) acc = fmaf(Wr[k], fS[k], acc);
    hp[t] = fmaxf(acc, 0.f);
  }
  __syncthreads();
  if (t < 48) {
    float acc = bc2[t];
    const float* Wr = Wc2 + (size_t)t * 128;
    for (int k = 0; k < 128; ++k) acc = fmaf(Wr[k], h1[k], acc);
    oc[t] = acc;
  }
  if (t >= 64 && t < 71) {
    int j = t - 64;
    float acc = bp2[j];
    const float* Wr = Wp2 + (size_t)j * 64;
    for (int k = 0; k < 64; ++k) acc = fmaf(Wr[k], hp[k], acc);
    float scale = (j < 3) ? 0.1f : ((j < 6) ? 1.57f : 0.05f);
    out[3520 + b * 7 + j] = scale * tanhf(acc);   // p_pred section
  }
  __syncthreads();
  if (t < 12) {
    float n0 = oc[4*t], n1 = oc[4*t+1], n2 = oc[4*t+2];
    float nn = sqrtf(n0*n0 + n1*n1 + n2*n2);
    float d = fmaxf(nn, 1e-12f);
    out[448 + b * 36 + 3*t + 0] = n0 / d;         // normals section
    out[448 + b * 36 + 3*t + 1] = n1 / d;
    out[448 + b * 36 + 3*t + 2] = n2 / d;
    float x = oc[4*t+3];
    float sp = (x > 20.f) ? x : log1pf(expf(x));  // softplus
    out[2752 + b * 12 + t] = sp + 0.005f;         // h section
  }
}

// K5: 500-iteration projected-gradient QP, one wave32 per batch element.
// Lanes 0..11 own lambda rows; M rows in registers, broadcasts via ds_bpermute.
__global__ __launch_bounds__(32) void k5_qp(float* __restrict__ out) {
  int b = blockIdx.x, c = threadIdx.x;
  bool act = (c < 12);
  float n0 = 0.f, n1 = 0.f, n2 = 0.f, hc = 0.f;
  if (act) {
    n0 = out[448 + b * 36 + 3*c + 0];
    n1 = out[448 + b * 36 + 3*c + 1];
    n2 = out[448 + b * 36 + 3*c + 2];
    hc = out[2752 + b * 12 + c];
  }
  float p0 = out[3520 + b*7 + 0], p1 = out[3520 + b*7 + 1], p2 = out[3520 + b*7 + 2];
  float Mrow[12];
  float diag = 0.f;
#pragma unroll
  for (int d = 0; d < 12; ++d) {
    float m = n0 * __shfl(n0, d) + n1 * __shfl(n1, d) + n2 * __shfl(n2, d);
    if (d == c) { m += 5.0e-4f; diag = m; }     // + I/(2*S_PEN)
    Mrow[d] = m;
  }
  float q = n0 * p0 + n1 * p1 + n2 * p2 - hc;
  float tr = 0.f;
#pragma unroll
  for (int d = 0; d < 12; ++d) tr += __shfl(diag, d);
  float alpha = 1.f / tr;
  float lam = 0.f;
  for (int it = 0; it < 500; ++it) {
    float g = -q;
#pragma unroll
    for (int d = 0; d < 12; ++d) g = fmaf(Mrow[d], __shfl(lam, d), g);
    lam = fmaxf(fmaf(-alpha, g, lam), 0.f);
  }
  float c0 = n0 * lam, c1 = n1 * lam, c2 = n2 * lam;   // zero for lanes >= 12
#pragma unroll
  for (int off = 16; off > 0; off >>= 1) {
    c0 += __shfl_xor(c0, off);
    c1 += __shfl_xor(c1, off);
    c2 += __shfl_xor(c2, off);
  }
  if (c == 0) {
    out[b*7 + 0] = p0 - c0;
    out[b*7 + 1] = p1 - c1;
    out[b*7 + 2] = p2 - c2;
    out[b*7 + 3] = out[3520 + b*7 + 3];
    out[b*7 + 4] = out[3520 + b*7 + 4];
    out[b*7 + 5] = out[3520 + b*7 + 5];
    out[b*7 + 6] = out[3520 + b*7 + 6];
  }
}

// ---------------------------------------------------------------------------
extern "C" void kernel_launch(void* const* d_in, const int* in_sizes, int n_in,
                              void* d_out, int out_size, void* d_ws, size_t ws_size,
                              hipStream_t stream) {
  (void)in_sizes; (void)n_in; (void)out_size; (void)ws_size;
  const float* pcd = (const float*)d_in[0];
  const float* W1  = (const float*)d_in[1];
  const float* b1  = (const float*)d_in[2];
  const float* g1  = (const float*)d_in[3];
  const float* be1 = (const float*)d_in[4];
  const float* W2  = (const float*)d_in[5];
  const float* b2  = (const float*)d_in[6];
  const float* g2  = (const float*)d_in[7];
  const float* be2 = (const float*)d_in[8];
  const float* W3  = (const float*)d_in[9];
  const float* b3  = (const float*)d_in[10];
  const float* g3  = (const float*)d_in[11];
  const float* be3 = (const float*)d_in[12];
  const float* Wfc = (const float*)d_in[13];
  const float* bfc = (const float*)d_in[14];
  const float* Wc1 = (const float*)d_in[15];
  const float* bc1 = (const float*)d_in[16];
  const float* Wc2 = (const float*)d_in[17];
  const float* bc2 = (const float*)d_in[18];
  const float* Wp1 = (const float*)d_in[19];
  const float* bp1 = (const float*)d_in[20];
  const float* Wp2 = (const float*)d_in[21];
  const float* bp2 = (const float*)d_in[22];
  float* out = (float*)d_out;

  float* ws = (float*)d_ws;
  size_t o = 0;
  float* pcd_part = ws + o; o += (size_t)NB_STAT * 9;
  float* s1 = ws + o; o += 64;
  float* t1 = ws + o; o += 64;
  float* S1part = ws + o; o += (size_t)NB_STAT * 4160;
  float* S1red  = ws + o; o += 4160;
  float* s2 = ws + o; o += 128;
  float* t2 = ws + o; o += 128;
  float* S2part = ws + o; o += (size_t)NB_STAT * 16512;
  float* S2red  = ws + o; o += 16512;
  float* s3 = ws + o; o += 256;
  float* t3 = ws + o; o += 256;
  float* maxp  = ws + o; o += (size_t)1024 * 256;
  float* feats = ws + o; o += (size_t)64 * 128;

  k0_pcd_stats<<<256, 256, 0, stream>>>(pcd, pcd_part);
  k0b_bn1<<<1, 64, 0, stream>>>(pcd_part, 256, W1, b1, g1, be1, s1, t1);
  k1_gram1<<<NB_STAT, 256, 0, stream>>>(pcd, W1, b1, s1, t1, S1part);
  k_reduce<<<(4160 + 255) / 256, 256, 0, stream>>>(S1part, S1red, 4160, NB_STAT);
  k1c_bn2<<<1, 128, 0, stream>>>(S1red, W2, b2, g2, be2, s2, t2);
  k2_gram2<<<NB_STAT, 256, 0, stream>>>(pcd, W1, b1, s1, t1, W2, b2, s2, t2, S2part);
  k_reduce<<<(16512 + 255) / 256, 256, 0, stream>>>(S2part, S2red, 16512, NB_STAT);
  k2c_bn3<<<1, 256, 0, stream>>>(S2red, W3, b3, g3, be3, s3, t3);
  k3_forward_max<<<1024, 256, 0, stream>>>(pcd, W1, b1, s1, t1, W2, b2, s2, t2,
                                           W3, b3, s3, t3, maxp);
  k3b_feats<<<64, 256, 0, stream>>>(maxp, Wfc, bfc, feats);
  k4_heads<<<64, 128, 0, stream>>>(feats, Wc1, bc1, Wc2, bc2, Wp1, bp1, Wp2, bp2, out);
  k5_qp<<<64, 32, 0, stream>>>(out);
}
